// triton_gemm_rs_intra_node_68925635166437
// MI455X (gfx1250) — matrix operand order repaired and checked by reading, not compiled, not run on hardware
//
#include <hip/hip_runtime.h>

// ---------------------------------------------------------------------------
// gemm_rs_intra_node on one GPU:
//   out[8192,4096] = sum_{src<8} a[src](8192x512) @ b[src](4096x512)^T
// Implemented as a single tiled GEMM (M=8192, N=4096, K=8*512) using
// bf16x3 splitting (hi/lo) + v_wmma_f32_16x16x32_bf16 with f32 accumulation.
// ---------------------------------------------------------------------------

#define NUM_RANKS 8
#define M_GLOBAL 8192
#define N_GLOBAL 4096
#define K_LOCAL  512
#define K_GLOBAL (NUM_RANKS * K_LOCAL)   // 4096

#define BM 128
#define BN 128
#define BK 32
#define LDST 40                 // padded LDS row stride in bf16 elements (80 B)
#define THREADS 256             // 8 wave32s

typedef __attribute__((ext_vector_type(8)))  float   v8f;
typedef __attribute__((ext_vector_type(8)))  __bf16  v8bf;
typedef __attribute__((ext_vector_type(16))) __bf16  v16bf;

// Build one 32-byte WMMA operand from two 16-byte LDS chunks.
__device__ __forceinline__ v16bf ld_frag(const __bf16* base, int o0, int o1) {
  v8bf x = *(const v8bf*)(base + o0);
  v8bf y = *(const v8bf*)(base + o1);
  return __builtin_shufflevector(x, y, 0, 1, 2, 3, 4, 5, 6, 7,
                                       8, 9, 10, 11, 12, 13, 14, 15);
}

__global__ __launch_bounds__(THREADS)
void gemm_rs_bf16x3_kernel(const float* __restrict__ A,   // [8, 8192, 512]
                           const float* __restrict__ B,   // [8, 4096, 512]
                           float* __restrict__ C) {       // [8192, 4096]
  __shared__ __bf16 sAh[BM * LDST];
  __shared__ __bf16 sAl[BM * LDST];
  __shared__ __bf16 sBh[BN * LDST];
  __shared__ __bf16 sBl[BN * LDST];

  const int tid  = threadIdx.x;
  const int lane = tid & 31;
  const int wid  = tid >> 5;        // 0..7
  const int wm   = wid >> 2;        // 0..1 : 64 output rows per wave
  const int wn   = wid & 3;         // 0..3 : 32 output cols per wave
  const int bm   = blockIdx.y * BM;
  const int bn   = blockIdx.x * BN;

  const int half = lane >> 4;       // lane group 0/1 (K-split per WMMA layout)
  const int lidx = lane & 15;       // M (A frag) or N (B frag) within 16

  // 8 f32 16x16 accumulators per wave (64 VGPRs)
  v8f acc[4][2];
#pragma unroll
  for (int i = 0; i < 4; ++i)
#pragma unroll
    for (int j = 0; j < 2; ++j)
#pragma unroll
      for (int e = 0; e < 8; ++e) acc[i][j][e] = 0.0f;

  // Staging map: 256 threads x float4 -> 32 rows x 32 cols per pass, 4 passes.
  const int scol  = (tid & 7) * 4;  // 0,4,...,28
  const int srow0 = tid >> 3;       // 0..31

  for (int kt = 0; kt < K_GLOBAL / BK; ++kt) {
    const int kg  = kt * BK;
    const int src = kg >> 9;               // kg / K_LOCAL (BK divides K_LOCAL)
    const int k0  = kg & (K_LOCAL - 1);

    const float* Asrc = A + (size_t)src * M_GLOBAL * K_LOCAL;
    const float* Bsrc = B + (size_t)src * N_GLOBAL * K_LOCAL;

    __syncthreads();  // previous compute done before LDS overwrite
#pragma unroll
    for (int p = 0; p < 4; ++p) {
      const int row = srow0 + p * 32;
      {
        const float4 v =
            *(const float4*)(Asrc + (size_t)(bm + row) * K_LOCAL + k0 + scol);
        const float f[4] = {v.x, v.y, v.z, v.w};
#pragma unroll
        for (int e = 0; e < 4; ++e) {
          const __bf16 h = (__bf16)f[e];
          const __bf16 l = (__bf16)(f[e] - (float)h);  // exact residual
          sAh[row * LDST + scol + e] = h;
          sAl[row * LDST + scol + e] = l;
        }
      }
      {
        const float4 v =
            *(const float4*)(Bsrc + (size_t)(bn + row) * K_LOCAL + k0 + scol);
        const float f[4] = {v.x, v.y, v.z, v.w};
#pragma unroll
        for (int e = 0; e < 4; ++e) {
          const __bf16 h = (__bf16)f[e];
          const __bf16 l = (__bf16)(f[e] - (float)h);
          sBh[row * LDST + scol + e] = h;
          sBl[row * LDST + scol + e] = l;
        }
      }
    }
    // Prefetch next K tile into the cache hierarchy (global_prefetch_b8).
    if (kt + 1 < K_GLOBAL / BK) {
      const int kg2  = kg + BK;
      const int src2 = kg2 >> 9;
      const int k02  = kg2 & (K_LOCAL - 1);
      __builtin_prefetch(A + (size_t)src2 * M_GLOBAL * K_LOCAL +
                             (size_t)(bm + srow0) * K_LOCAL + k02 + scol, 0, 0);
      __builtin_prefetch(B + (size_t)src2 * N_GLOBAL * K_LOCAL +
                             (size_t)(bn + srow0) * K_LOCAL + k02 + scol, 0, 0);
    }
    __syncthreads();

    // B operands (32x16, 16-bit): lane = N (lidx), halves hold K 0..15 / 16..31
    v16bf bh[2], bl[2];
#pragma unroll
    for (int tn = 0; tn < 2; ++tn) {
      const int n = wn * 32 + tn * 16 + lidx;
      const int o = n * LDST + half * 16;
      bh[tn] = ld_frag(sBh, o, o + 8);
      bl[tn] = ld_frag(sBl, o, o + 8);
    }

#pragma unroll
    for (int tm = 0; tm < 4; ++tm) {
      // A operand (16x32, 16-bit): lane = M (lidx); half selects K 0-15
      // vs 16-31 (same K order as B).
      const int m = wm * 64 + tm * 16 + lidx;
      const int o = m * LDST + half * 16;
      const v16bf ah = ld_frag(sAh, o, o + 8);
      const v16bf al = ld_frag(sAl, o, o + 8);
#pragma unroll
      for (int tn = 0; tn < 2; ++tn) {
        // bf16x3: hi*hi + hi*lo + lo*hi (lo*lo ~ 2^-32, dropped)
        acc[tm][tn] = __builtin_amdgcn_wmma_f32_16x16x32_bf16(
            false, ah, false, bh[tn], (short)0, acc[tm][tn], false, false);
        acc[tm][tn] = __builtin_amdgcn_wmma_f32_16x16x32_bf16(
            false, ah, false, bl[tn], (short)0, acc[tm][tn], false, false);
        acc[tm][tn] = __builtin_amdgcn_wmma_f32_16x16x32_bf16(
            false, al, false, bh[tn], (short)0, acc[tm][tn], false, false);
      }
    }
  }

  // Write out: f32 16x16 C layout — VGPR g: M = g + 8*half, N = lidx.
#pragma unroll
  for (int tm = 0; tm < 4; ++tm) {
    const int row_base = bm + wm * 64 + tm * 16 + half * 8;
#pragma unroll
    for (int tn = 0; tn < 2; ++tn) {
      const int col = bn + wn * 32 + tn * 16 + lidx;
#pragma unroll
      for (int g = 0; g < 8; ++g) {
        C[(size_t)(row_base + g) * N_GLOBAL + col] = acc[tm][tn][g];
      }
    }
  }
}

extern "C" void kernel_launch(void* const* d_in, const int* in_sizes, int n_in,
                              void* d_out, int out_size, void* d_ws, size_t ws_size,
                              hipStream_t stream) {
  const float* a = (const float*)d_in[0];   // [8, 8192, 512] f32
  const float* b = (const float*)d_in[1];   // [8, 4096, 512] f32
  float* out     = (float*)d_out;           // [8, 1024, 4096] == [8192, 4096] f32

  dim3 grid(N_GLOBAL / BN, M_GLOBAL / BM);  // 32 x 64 = 2048 workgroups
  dim3 block(THREADS);
  gemm_rs_bf16x3_kernel<<<grid, block, 0, stream>>>(a, b, out);
}
